// WindowAttention_59064390255030
// MI455X (gfx1250) — matrix operand order repaired and checked by reading, not compiled, not run on hardware
//
#include <hip/hip_runtime.h>

// ---------- CDNA5 wave32 WMMA types ----------
typedef __bf16 bf16;
typedef __attribute__((ext_vector_type(16))) __bf16 v16bf;
typedef __attribute__((ext_vector_type(8)))  float  v8f;

#define HEADS   3
#define C_DIM   96
#define QKV_DIM 288
#define N_TOK   98      // 2*7*7 tokens per window
#define N_PAD   112     // padded to 7 WMMA M-tiles
#define K_PAD   128     // PV K padded to 4 chunks of 32
#define SCALE_Q 0.17677669529663687f  // 32^-0.5

// ---------- LDS region plan (bytes), unioned across phases ----------
// A: qkv_w bf16 [288][96]=55296 | scores f32 [112][112]=50176 | proj_w bf16 [96][96]=18432
// B: sX bf16 [112][96]=21504    | probs bf16 [112][128]=28672
// C: sQKV bf16 [112][288]=64512
// D: sO bf16 [112][96]=21504
// E: sVt bf16 [96][128]=24576   (V transposed, K-padded; contiguous PV B-fragments)
// F: bias table f32 507*3=6084 (-> 6096)
#define OFF_A 0
#define OFF_B 55296
#define OFF_C 83968
#define OFF_D 148480
#define OFF_E 169984
#define OFF_F 194560
#define SMEM_BYTES 200656

// ---------------- fragment loaders (contiguous-k layouts, wave32) ----------------
// A 16x32 bf16: lanes 0-15 -> M=lane, K=kb..kb+15, kb = (lane<16?0:16)
__device__ __forceinline__ v16bf frag_a(const bf16* m, int ldm, int row, int kOff, int kbase) {
  v16bf a;
  const bf16* p = m + row * ldm + kOff + kbase;
#pragma unroll
  for (int i = 0; i < 16; ++i) a[i] = p[i];
  return a;
}
// B 32x16 bf16 with B(k,n) = src[n*ldm + k] (i.e. B = srcT, contiguous in k):
// lanes 0-15 -> N=lane, K=0..15 ; lanes 16-31 -> N=lane-16, K=16..31
__device__ __forceinline__ v16bf frag_bT(const bf16* m, int ldm, int col, int kOff, int kb) {
  v16bf b;
  const bf16* p = m + col * ldm + kOff + kb;
#pragma unroll
  for (int i = 0; i < 16; ++i) b[i] = p[i];
  return b;
}
__device__ __forceinline__ v8f wmma_bf16(v16bf a, v16bf b, v8f c) {
  return __builtin_amdgcn_wmma_f32_16x16x32_bf16(false, a, false, b, (short)0, c, false, false);
}

// token n in window (2,7,7) -> rolled source linear index within one batch image
__device__ __forceinline__ int token_src_lin(int n, int sB, int hB, int wB) {
  int t0 = n / 49;
  int r  = n - t0 * 49;
  int t1 = r / 7;
  int t2 = r - t1 * 7;
  int ss = (sB + t0 + 1) & 7;              // roll by tshift=(1,6,6), Sp=8
  int hs = hB + t1 + 6; if (hs >= 112) hs -= 112;
  int ws = wB + t2 + 6; if (ws >= 112) ws -= 112;
  return (ss * 112 + hs) * 112 + ws;
}

// relative position bias from LDS-staged table: idx = ((dz+1)*13 + (dy+6))*13 + (dx+6)
__device__ __forceinline__ float rel_bias(const float* sT, int mg, int ng, int h) {
  int a0 = mg / 49, ra = mg - a0 * 49, a1 = ra / 7, a2 = ra - a1 * 7;
  int b0 = ng / 49, rb = ng - b0 * 49, b1 = rb / 7, b2 = rb - b1 * 7;
  int idx = ((a0 - b0 + 1) * 13 + (a1 - b1 + 6)) * 13 + (a2 - b2 + 6);
  return sT[idx * HEADS + h];
}

__global__ __launch_bounds__(256) void win_attn_fused(
    const float* __restrict__ x,       // (4, 8*112*112, 96)
    const float* __restrict__ qkv_w,   // (288, 96)
    const float* __restrict__ qkv_b,   // (288,)
    const float* __restrict__ proj_w,  // (96, 96)
    const float* __restrict__ proj_b,  // (96,)
    const float* __restrict__ tbl,     // (507, 3)
    float* __restrict__ out)           // (4, 8*112*112, 96)
{
  extern __shared__ char smem[];
  bf16*  sWq  = (bf16*) (smem + OFF_A);
  float* sS   = (float*)(smem + OFF_A);
  bf16*  sPW  = (bf16*) (smem + OFF_A);
  bf16*  sX   = (bf16*) (smem + OFF_B);
  bf16*  sP   = (bf16*) (smem + OFF_B);
  bf16*  sQKV = (bf16*) (smem + OFF_C);
  bf16*  sO   = (bf16*) (smem + OFF_D);
  bf16*  sVt  = (bf16*) (smem + OFF_E);
  float* sT   = (float*)(smem + OFF_F);

  const int tid   = threadIdx.x;
  // wave id is genuinely wave-uniform: force it scalar so all tile bookkeeping
  // (t/mt/nt, loop branches, the V-store test) lives in SGPRs, not EXEC masks.
  const int wave  = __builtin_amdgcn_readfirstlane(tid >> 5);
  const int lane  = tid & 31;
  const int lr    = lane & 15;
  const int hi    = lane >> 4;
  const int kbase = hi * 16;  // A-fragment K base
  const int kb    = hi * 16;  // B-fragment K base

  // decode window id: flatten order (b, i1, i2, s', h', w') with dil0=1
  const int wid = blockIdx.x;
  const int wq = wid & 7;
  const int hq = (wid >> 3) & 7;
  const int sq = (wid >> 6) & 3;
  const int i2 = (wid >> 8) & 1;
  const int i1 = (wid >> 9) & 1;
  const int bb = wid >> 10;
  const int sB = sq * 2;
  const int hB = hq * 14 + i1 * 7;
  const int wB = wq * 14 + i2 * 7;

  const float* xb = x   + (size_t)bb * (8 * 112 * 112) * C_DIM;
  float*       ob = out + (size_t)bb * (8 * 112 * 112) * C_DIM;

  __builtin_prefetch(qkv_w, 0, 3);   // global_prefetch_b8: pull weights toward caches
  __builtin_prefetch(proj_w, 0, 3);

  // ---------------- Phase 0: gather window + stage weights/bias (fp32 -> bf16) ----------------
  for (int e = tid; e < N_TOK * 24; e += 256) {            // 24 float4 per 96-ch token
    int n = e / 24, j = e - n * 24;
    int lin = token_src_lin(n, sB, hB, wB);
    float4 v = ((const float4*)(xb + (size_t)lin * C_DIM))[j];
    int o = n * C_DIM + j * 4;
    sX[o + 0] = (bf16)v.x; sX[o + 1] = (bf16)v.y;
    sX[o + 2] = (bf16)v.z; sX[o + 3] = (bf16)v.w;
  }
  for (int e = tid; e < (N_PAD - N_TOK) * C_DIM; e += 256) // zero pad rows 98..111
    sX[N_TOK * C_DIM + e] = (bf16)0.0f;
  for (int e = tid; e < QKV_DIM * 24; e += 256) {          // qkv_w copy (row-major match)
    float4 v = ((const float4*)qkv_w)[e];
    int o = e * 4;
    sWq[o + 0] = (bf16)v.x; sWq[o + 1] = (bf16)v.y;
    sWq[o + 2] = (bf16)v.z; sWq[o + 3] = (bf16)v.w;
  }
  for (int e = tid; e < C_DIM * 16; e += 256) {            // zero sVt K-pad cols 112..127
    int n = e >> 4, k = N_PAD + (e & 15);
    sVt[n * K_PAD + k] = (bf16)0.0f;
  }
  for (int e = tid; e < 507 * HEADS; e += 256)             // stage rel-bias table
    sT[e] = tbl[e];
  __syncthreads();

  // ---------------- Phase 1: QKV = X @ WqT + b, fused Q-scale; V also stored transposed ----------------
  for (int t = wave; t < 7 * 18; t += 8) {
    int mt = t / 18, nt = t - mt * 18;                      // scalar (t uniform)
    v8f acc = {};
#pragma unroll
    for (int kc = 0; kc < 3; ++kc) {
      v16bf a = frag_a (sX,  C_DIM, mt * 16 + lr, kc * 32, kbase);
      v16bf b = frag_bT(sWq, C_DIM, nt * 16 + lr, kc * 32, kb);
      acc = wmma_bf16(a, b, acc);
    }
    int colg = nt * 16 + lr;
    float bia = qkv_b[colg];
    float sc  = (nt < 6) ? SCALE_Q : 1.0f;                  // Q cols are tiles 0..5 (uniform)
    bf16* vtCol = sVt + (colg - 2 * C_DIM) * K_PAD + mt * 16 + hi * 8; // 8 consecutive rows
    if (nt >= 12) {                                         // V tiles: uniform scalar branch
#pragma unroll
      for (int v = 0; v < 8; ++v) {
        int row = mt * 16 + hi * 8 + v;
        bf16 bval = (bf16)(acc[v] + bia);
        sQKV[row * QKV_DIM + colg] = bval;
        vtCol[v] = bval;                                    // contiguous transposed V store
      }
    } else {
#pragma unroll
      for (int v = 0; v < 8; ++v) {
        int row = mt * 16 + hi * 8 + v;                     // D layout: m = hi*8+vgpr, n = lr
        sQKV[row * QKV_DIM + colg] = (bf16)((acc[v] + bia) * sc);
      }
    }
  }
  __syncthreads();

  // ---------------- Phase 2: per-head attention ----------------
  for (int h = 0; h < HEADS; ++h) {
    const int qo = h * 32, ko = C_DIM + h * 32;

    // 2a: S = Q KT + relbias (masked), 7x7 tiles, K=32
    for (int t = wave; t < 49; t += 8) {
      int mt = t / 7, nt = t - mt * 7;
      v16bf a = frag_a (sQKV, QKV_DIM, mt * 16 + lr, qo, kbase);
      v16bf b = frag_bT(sQKV, QKV_DIM, nt * 16 + lr, ko, kb);  // B(k,n)=K[n][k]
      v8f acc = {};
      acc = wmma_bf16(a, b, acc);
      int ng = nt * 16 + lr;
#pragma unroll
      for (int v = 0; v < 8; ++v) {
        int mg = mt * 16 + hi * 8 + v;
        float val = (mg < N_TOK && ng < N_TOK) ? (acc[v] + rel_bias(sT, mg, ng, h))
                                               : -1.0e30f;
        sS[mg * N_PAD + ng] = val;
      }
    }
    __syncthreads();

    // 2b: wave-parallel row softmax; lanes 16-31's 4th column lands on K-pad cols (writes 0)
    for (int r = wave; r < N_PAD; r += 8) {
      const float* rp = sS + r * N_PAD;
      float a0 = rp[lane];
      float a1 = rp[lane + 32];
      float a2 = rp[lane + 64];
      float a3 = (lane < 16) ? rp[lane + 96] : -3.4e38f;
      float mx = fmaxf(fmaxf(a0, a1), fmaxf(a2, a3));
#pragma unroll
      for (int off = 16; off > 0; off >>= 1) mx = fmaxf(mx, __shfl_xor(mx, off, 32));
      float e0 = __expf(a0 - mx);
      float e1 = __expf(a1 - mx);
      float e2 = __expf(a2 - mx);
      float e3 = (lane < 16) ? __expf(a3 - mx) : 0.0f;
      float sum = e0 + e1 + e2 + e3;
#pragma unroll
      for (int off = 16; off > 0; off >>= 1) sum += __shfl_xor(sum, off, 32);
      float inv = 1.0f / sum;
      bf16* pp = sP + r * K_PAD;
      pp[lane]      = (bf16)(e0 * inv);
      pp[lane + 32] = (bf16)(e1 * inv);
      pp[lane + 64] = (bf16)(e2 * inv);
      pp[lane + 96] = (bf16)(e3 * inv);   // lanes>=16 -> cols 112..127 = 0 (K-pad)
    }
    __syncthreads();

    // 2c: O_h = P @ V_h (M=112, N=32, K=128 padded); B from transposed V, contiguous
    for (int t = wave; t < 14; t += 8) {
      int mt = t >> 1, nt = t & 1;
      v8f acc = {};
#pragma unroll
      for (int kc = 0; kc < 4; ++kc) {
        v16bf a = frag_a (sP,  K_PAD, mt * 16 + lr, kc * 32, kbase);
        v16bf b = frag_bT(sVt, K_PAD, h * 32 + nt * 16 + lr, kc * 32, kb); // B(k,n)=sVt[n][k]
        acc = wmma_bf16(a, b, acc);
      }
#pragma unroll
      for (int v = 0; v < 8; ++v) {
        int row = mt * 16 + hi * 8 + v;
        sO[row * C_DIM + h * 32 + nt * 16 + lr] = (bf16)acc[v];
      }
    }
    __syncthreads();
  }

  // ---------------- Phase 3: proj = O @ WpT + b, scatter to un-rolled positions ----------------
  for (int e = tid; e < C_DIM * 24; e += 256) {
    float4 v = ((const float4*)proj_w)[e];
    int o = e * 4;
    sPW[o + 0] = (bf16)v.x; sPW[o + 1] = (bf16)v.y;
    sPW[o + 2] = (bf16)v.z; sPW[o + 3] = (bf16)v.w;
  }
  __syncthreads();

  for (int t = wave; t < 7 * 6; t += 8) {
    int mt = t / 6, nt = t - mt * 6;
    v8f acc = {};
#pragma unroll
    for (int kc = 0; kc < 3; ++kc) {
      v16bf a = frag_a (sO,  C_DIM, mt * 16 + lr, kc * 32, kbase);
      v16bf b = frag_bT(sPW, C_DIM, nt * 16 + lr, kc * 32, kb);
      acc = wmma_bf16(a, b, acc);
    }
    int colg = nt * 16 + lr;
    float pb = proj_b[colg];
#pragma unroll
    for (int v = 0; v < 8; ++v) {
      int mg = mt * 16 + hi * 8 + v;
      if (mg < N_TOK) {
        int lin = token_src_lin(mg, sB, hB, wB);   // reverse-partition + roll-back is same map
        ob[(size_t)lin * C_DIM + colg] = acc[v] + pb;
      }
    }
  }
}

extern "C" void kernel_launch(void* const* d_in, const int* in_sizes, int n_in,
                              void* d_out, int out_size, void* d_ws, size_t ws_size,
                              hipStream_t stream) {
  (void)in_sizes; (void)n_in; (void)out_size; (void)d_ws; (void)ws_size;
  const float* x      = (const float*)d_in[0];
  const float* qkv_w  = (const float*)d_in[1];
  const float* qkv_b  = (const float*)d_in[2];
  const float* proj_w = (const float*)d_in[3];
  const float* proj_b = (const float*)d_in[4];
  const float* tbl    = (const float*)d_in[5];
  float* out = (float*)d_out;

  (void)hipFuncSetAttribute((const void*)win_attn_fused,
                            hipFuncAttributeMaxDynamicSharedMemorySize, SMEM_BYTES);

  // 4096 windows, one workgroup (8 wave32) each; all GEMM work via v_wmma_f32_16x16x32_bf16
  win_attn_fused<<<dim3(4096), dim3(256), SMEM_BYTES, stream>>>(
      x, qkv_w, qkv_b, proj_w, proj_b, tbl, out);
}
